// BahdanauAttention_57062935494893
// MI455X (gfx1250) — matrix operand order repaired and checked by reading, not compiled, not run on hardware
//
#include <hip/hip_runtime.h>

static constexpr int kS = 2048;
static constexpr int kB = 32;
static constexpr int kH = 1024;

typedef __attribute__((ext_vector_type(16))) __bf16 v16bf;
typedef __attribute__((ext_vector_type(8)))  __bf16 v8bf;
typedef __attribute__((ext_vector_type(8)))  float  v8f;

// Branchless tanh: sign(x)*(1-e^-2|x|)/(1+e^-2|x|). v_exp_f32 underflow -> +-1
// saturation for free. ~6 straight-line VALU ops, no EXEC divergence (the OCML
// tanhf emitted v_cmpx + branches inside the hot epilogue).
__device__ __forceinline__ float fast_tanh(float x) {
    const float ax = __builtin_fabsf(x);
    const float e  = __builtin_amdgcn_exp2f(ax * -2.885390082f); // exp(-2ax)
    const float t  = (1.0f - e) * __builtin_amdgcn_rcpf(1.0f + e);
    return __builtin_copysignf(t, x);
}

// ---------------------------------------------------------------------------
// Kernel 1: W1 (f32) -> bf16; 2 MB result stays hot in 192 MB L2 for the GEMM.
// ---------------------------------------------------------------------------
__global__ void k_w1_to_bf16(const float* __restrict__ W1,
                             __bf16* __restrict__ W1bf) {
    const int i = (blockIdx.x * blockDim.x + threadIdx.x) * 4;
    const float4 f = *(const float4*)(W1 + i);
    W1bf[i + 0] = (__bf16)f.x;
    W1bf[i + 1] = (__bf16)f.y;
    W1bf[i + 2] = (__bf16)f.z;
    W1bf[i + 3] = (__bf16)f.w;
}

// ---------------------------------------------------------------------------
// Kernel 2: hb[b][k] = sum_h hidden[b,h]*W2[k,h] + b1[k] + b2[k]  (once).
// ---------------------------------------------------------------------------
__global__ void k_hidden_proj(const float* __restrict__ hidden,
                              const float* __restrict__ W2,
                              const float* __restrict__ b1,
                              const float* __restrict__ b2,
                              float* __restrict__ hb) {
    __shared__ float hrow[kH];
    const int b = blockIdx.x;
    for (int h = threadIdx.x; h < kH; h += blockDim.x)
        hrow[h] = hidden[b * kH + h];
    __syncthreads();
    for (int k = threadIdx.x; k < kH; k += blockDim.x) {
        const float* wr = W2 + (size_t)k * kH;
        float acc = 0.f;
        for (int h = 0; h < kH; h += 4) {
            const float4 w = *(const float4*)(wr + h);
            acc += w.x * hrow[h] + w.y * hrow[h + 1] +
                   w.z * hrow[h + 2] + w.w * hrow[h + 3];
        }
        hb[b * kH + k] = acc + b1[k] + b2[k];
    }
}

// ---------------------------------------------------------------------------
// Kernel 3 (main): energy[r] = v . tanh(hs[r,:] @ W1^T + hb[r%32,:])
// Block = 16 rows (one WMMA M-tile); 4 waves split N=1024 into 4x256.
// A tile (16 rows = one contiguous 64 KB region) is copied to LDS with
// gfx1250 global_load_async_to_lds_b128 (ASYNCcnt), then converted f32->bf16
// in LDS. K=1024 accumulated with v_wmma_f32_16x16x32_bf16; epilogue fuses
// bias + branchless tanh + v-dot so `pre` (256 MB) never touches memory.
// Dynamic LDS: [stage f32 64 KB][As bf16 16x1032 = 33 KB][eSh 64 B].
// ---------------------------------------------------------------------------
__global__ void __launch_bounds__(128)
k_energy(const float* __restrict__ hs, const __bf16* __restrict__ W1bf,
         const float* __restrict__ hb, const float* __restrict__ v,
         const int* __restrict__ mask, float* __restrict__ E) {
    extern __shared__ char smem[];
    float* stage = (float*)smem;                                  // 64 KB
    __bf16 (*As)[kH + 8] = (__bf16(*)[kH + 8])(smem + 64 * 1024); // 33 KB
    float* eSh = (float*)(smem + 64 * 1024 + 16 * (kH + 8) * 2);

    const int tid = threadIdx.x;
    const int rowBase = blockIdx.x * 16;

    if (tid < 16) eSh[tid] = 0.0f;

    // ---- Stage 1: async bulk copy, global -> LDS (no VGPR round trip) ----
    {
        const unsigned ldsBase = (unsigned)(size_t)(void*)stage;
        const float* gsrc = hs + (size_t)rowBase * kH;
        for (int i = tid; i < (16 * kH) / 4; i += 128) {   // float4 granules
            const unsigned lds = ldsBase + (unsigned)i * 16u;
            const float* g = gsrc + i * 4;
            asm volatile("global_load_async_to_lds_b128 %0, %1, off"
                         :: "v"(lds), "v"(g) : "memory");
        }
        asm volatile("s_wait_asynccnt 0x0" ::: "memory");
    }
    __syncthreads();   // all waves' async copies complete

    // ---- Stage 2: LDS f32 -> LDS bf16 (padded rows, conflict-free b128) ----
    for (int idx = tid; idx < 16 * 256; idx += 128) {
        const int r  = idx >> 8;
        const int c4 = (idx & 255) << 2;
        const float4 f = *(const float4*)(stage + r * kH + c4);
        As[r][c4 + 0] = (__bf16)f.x;
        As[r][c4 + 1] = (__bf16)f.y;
        As[r][c4 + 2] = (__bf16)f.z;
        As[r][c4 + 3] = (__bf16)f.w;
    }
    __syncthreads();

    const int lane  = tid & 31;
    const int wave  = tid >> 5;
    const int nBase = wave * 256;          // this wave's N slice
    const int mlo   = lane & 15;           // A row / B,C column within tile
    const int hi    = lane >> 4;           // lane half select
    const int aKOff = hi << 4;             // A: K sub-offset (0 or 16)
    const int bKOff = hi << 4;             // B: K sub-offset (0 or 16)

    float eAcc[8] = {0.f, 0.f, 0.f, 0.f, 0.f, 0.f, 0.f, 0.f};

    for (int nt = 0; nt < 16; ++nt) {
        const int ncol = nBase + nt * 16 + mlo;
        const __bf16* brow = W1bf + (size_t)ncol * kH + bKOff;
        __builtin_prefetch(brow, 0, 0);    // global_prefetch_b8 (speculative)

        v8f c = {};
        #pragma unroll 4
        for (int kb = 0; kb < kH; kb += 32) {
            // A fragment: lane half -> K{aKOff..+7} and K{8+aKOff..+7}
            const v8bf alo = *(const v8bf*)(&As[mlo][kb + aKOff]);
            const v8bf ahi = *(const v8bf*)(&As[mlo][kb + 8 + aKOff]);
            v16bf a;
            #pragma unroll
            for (int e = 0; e < 8; ++e) { a[e] = alo[e]; a[e + 8] = ahi[e]; }
            // B fragment: lane -> column ncol, 16 contiguous K (32 bytes)
            const v16bf b = *(const v16bf*)(brow + kb);
            c = __builtin_amdgcn_wmma_f32_16x16x32_bf16(
                    false, a, false, b, (short)0, c, false, false);
        }

        // Fused epilogue: pre = c + hb ; e += tanh(pre) * v[n]  (branchless)
        const float vv = v[ncol];
        #pragma unroll
        for (int r = 0; r < 8; ++r) {
            const int m    = r + (hi << 3);            // row within tile
            const int bidx = (rowBase + m) & (kB - 1); // batch = row % 32
            const float pre = c[r] + hb[bidx * kH + ncol];
            eAcc[r] += fast_tanh(pre) * vv;
        }
    }

    // Reduce partial energies across the 16 columns held by each lane group.
    #pragma unroll
    for (int r = 0; r < 8; ++r) {
        float e = eAcc[r];
        for (int off = 8; off; off >>= 1) e += __shfl_xor(e, off, 16);
        if (mlo == 0) atomicAdd(&eSh[r + (hi << 3)], e);  // ds_add_f32
    }
    __syncthreads();

    if (tid < 16) {
        const int row = rowBase + tid;
        E[row] = mask[row] ? eSh[tid] : -1.0e10f;
    }
}

// ---------------------------------------------------------------------------
// Kernel 4: masked softmax over S per batch column, in place.
// ---------------------------------------------------------------------------
__global__ void k_softmax(float* __restrict__ E) {
    const int b = blockIdx.x;
    const int t = threadIdx.x;
    __shared__ float red[256];
    __shared__ float s_max, s_sum;

    float m = -3.0e38f;
    for (int s = t; s < kS; s += 256) m = fmaxf(m, E[s * kB + b]);
    red[t] = m;
    __syncthreads();
    for (int off = 128; off; off >>= 1) {
        if (t < off) red[t] = fmaxf(red[t], red[t + off]);
        __syncthreads();
    }
    if (t == 0) s_max = red[0];
    __syncthreads();
    m = s_max;

    float sum = 0.f;
    for (int s = t; s < kS; s += 256) sum += __expf(E[s * kB + b] - m);
    red[t] = sum;
    __syncthreads();
    for (int off = 128; off; off >>= 1) {
        if (t < off) red[t] += red[t + off];
        __syncthreads();
    }
    if (t == 0) s_sum = red[0];
    __syncthreads();

    const float inv = 1.0f / s_sum;
    for (int s = t; s < kS; s += 256)
        E[s * kB + b] = __expf(E[s * kB + b] - m) * inv;
}

// ---------------------------------------------------------------------------
// Kernel 5: zero the output (harness poisons d_out with 0xAA).
// ---------------------------------------------------------------------------
__global__ void k_zero(float* __restrict__ out) {
    out[blockIdx.x * blockDim.x + threadIdx.x] = 0.0f;
}

// ---------------------------------------------------------------------------
// Kernel 6: context[b,h] = sum_s attn[s,b]*hs[s,b,h]; S split 8 ways,
// partials combined with global_atomic_add_f32. Single coalesced 256 MB pass.
// ---------------------------------------------------------------------------
__global__ void k_context(const float* __restrict__ E,
                          const float* __restrict__ hs,
                          float* __restrict__ out) {
    const int b     = blockIdx.x & (kB - 1);
    const int chunk = blockIdx.x >> 5;
    const int h4    = threadIdx.x << 2;
    const int s0    = chunk * (kS / 8);

    float4 acc = make_float4(0.f, 0.f, 0.f, 0.f);
    for (int s = s0; s < s0 + kS / 8; ++s) {
        const float  w = E[s * kB + b];
        const float4 x = *(const float4*)(hs + (size_t)(s * kB + b) * kH + h4);
        acc.x += w * x.x; acc.y += w * x.y; acc.z += w * x.z; acc.w += w * x.w;
    }
    float* o = out + b * kH + h4;
    atomicAdd(o + 0, acc.x);
    atomicAdd(o + 1, acc.y);
    atomicAdd(o + 2, acc.z);
    atomicAdd(o + 3, acc.w);
}

// ---------------------------------------------------------------------------
extern "C" void kernel_launch(void* const* d_in, const int* in_sizes, int n_in,
                              void* d_out, int out_size, void* d_ws, size_t ws_size,
                              hipStream_t stream) {
    const float* hidden = (const float*)d_in[0];
    const float* hs     = (const float*)d_in[1];
    const int*   mask   = (const int*)  d_in[2];
    const float* W1     = (const float*)d_in[3];
    const float* b1     = (const float*)d_in[4];
    const float* W2     = (const float*)d_in[5];
    const float* b2     = (const float*)d_in[6];
    const float* v      = (const float*)d_in[7];
    float*       out    = (float*)d_out;

    // Workspace layout: [W1 bf16 : 2 MB][hb : 128 KB][energy/attn : 256 KB]
    char*   ws    = (char*)d_ws;
    __bf16* W1bf  = (__bf16*)ws;
    float*  hb    = (float*)(ws + (size_t)2 * 1024 * 1024);
    float*  E     = (float*)(ws + (size_t)2 * 1024 * 1024 + 128 * 1024);

    // Dynamic LDS for k_energy: 64 KB f32 stage + 16*(kH+8)*2 B bf16 + 64 B.
    const size_t dynLds = 64 * 1024 + (size_t)16 * (kH + 8) * 2 + 64;

    k_w1_to_bf16 <<<(kH * kH) / (256 * 4), 256, 0, stream>>>(W1, W1bf);
    k_hidden_proj<<<kB, 256, 0, stream>>>(hidden, W2, b1, b2, hb);
    k_energy     <<<(kS * kB) / 16, 128, dynLds, stream>>>(hs, W1bf, hb, v, mask, E);
    k_softmax    <<<kB, 256, 0, stream>>>(E);
    k_zero       <<<(kB * kH) / 256, 256, 0, stream>>>(out);
    k_context    <<<kB * 8, 256, 0, stream>>>(E, hs, out);
}
